// LiftSplatShoot_64965675319639
// MI455X (gfx1250) — matrix operand order repaired and checked by reading, not compiled, not run on hardware
//
#include <hip/hip_runtime.h>
#include <hip/hip_bf16.h>
#include <math.h>

#define B_    32
#define D_    118
#define HW576 576
#define NX_   256
#define NY_   256
#define NYOUT 206

typedef __attribute__((ext_vector_type(16))) _Float16 v16h;
typedef __attribute__((ext_vector_type(8)))  _Float16 v8h;
typedef __attribute__((ext_vector_type(8)))  float    v8f;

// ---------------------------------------------------------------- utilities
__global__ void k_zero_f32(float* p, long n) {
  long i = (long)blockIdx.x * blockDim.x + threadIdx.x;
  long s = (long)gridDim.x * blockDim.x;
  for (; i < n; i += s) p[i] = 0.f;
}

// ------------------------------------------------- geometry: combine = R@K^-1
__global__ void k_combine(const float* __restrict__ intr, float* __restrict__ comb) {
  int b = threadIdx.x;
  if (b >= B_) return;
  const float* K = intr + b * 9;
  float a = K[0], bb = K[1], c = K[2], d = K[3], e = K[4], f = K[5], g = K[6], h = K[7], i = K[8];
  float A = e * i - f * h, Bm = -(d * i - f * g), C = d * h - e * g;
  float det = a * A + bb * Bm + c * C;
  float inv[9];
  inv[0] = A / det;  inv[1] = -(bb * i - c * h) / det; inv[2] = (bb * f - c * e) / det;
  inv[3] = Bm / det; inv[4] = (a * i - c * g) / det;   inv[5] = -(a * f - c * d) / det;
  inv[6] = C / det;  inv[7] = -(a * h - bb * g) / det; inv[8] = (a * e - bb * d) / det;
  float* o = comb + b * 9;   // R = [[1,0,0],[0,0,1],[0,-1,0]]
  o[0] = inv[0];  o[1] = inv[1];  o[2] = inv[2];
  o[3] = inv[6];  o[4] = inv[7];  o[5] = inv[8];
  o[6] = -inv[3]; o[7] = -inv[4]; o[8] = -inv[5];
}

// ------------------------------------------------- voxel index per frustum pt
__global__ void k_voxel(const float* __restrict__ comb, int* __restrict__ vidx) {
  long i = (long)blockIdx.x * blockDim.x + threadIdx.x;
  long n = (long)B_ * D_ * HW576;
  if (i >= n) return;
  int hw = (int)(i % HW576);
  int d  = (int)((i / HW576) % D_);
  int b  = (int)(i / ((long)HW576 * D_));
  int h = hw / 24, w = hw % 24;
  float u = w * (383.0f / 23.0f), v = h * (383.0f / 23.0f);
  float dep = 1.0f + 0.5f * d;
  float p0 = u * dep, p1 = v * dep, p2 = dep;
  const float* cb = comb + b * 9;
  float gx = cb[0] * p0 + cb[1] * p1 + cb[2] * p2;
  float gy = cb[3] * p0 + cb[4] * p1 + cb[5] * p2;
  float gz = cb[6] * p0 + cb[7] * p1 + cb[8] * p2;
  int vx = (int)((gx + 51.2f) * 2.5f);   // trunc-toward-zero == .astype(int32)
  int vy = (int)((gy + 51.2f) * 2.5f);
  int vz = (int)((gz + 10.0f) * 0.05f);
  bool kept = (vx >= 0) && (vx < NX_) && (vy >= 0) && (vy < NY_) && (vz == 0);
  vidx[i] = kept ? (vx * NY_ + vy) : -1;
}

// ------------------------------------------------- depth_x global mean / var
__global__ void k_stat0(const float* __restrict__ x, long n, float* __restrict__ acc) {
  __shared__ float ss[256], sq[256];
  float s = 0.f, q = 0.f;
  for (long i = (long)blockIdx.x * blockDim.x + threadIdx.x; i < n;
       i += (long)gridDim.x * blockDim.x) {
    float v = x[i]; s += v; q += v * v;
  }
  ss[threadIdx.x] = s; sq[threadIdx.x] = q; __syncthreads();
  for (int o = 128; o > 0; o >>= 1) {
    if (threadIdx.x < o) { ss[threadIdx.x] += ss[threadIdx.x + o]; sq[threadIdx.x] += sq[threadIdx.x + o]; }
    __syncthreads();
  }
  if (threadIdx.x == 0) { atomicAdd(&acc[0], ss[0]); atomicAdd(&acc[1], sq[0]); }
}

// conv1(1x1)+BN fold:  t1 = relu(A_c * x + C_c)
__global__ void k_aff1(const float* __restrict__ acc, const float* __restrict__ w1,
                       const float* __restrict__ g1, const float* __restrict__ be1,
                       float* __restrict__ aff) {
  int c = threadIdx.x;
  if (c >= 8) return;
  float N = (float)B_ * 384.f * 384.f;
  float mx = acc[0] / N;
  float vx = acc[1] / N - mx * mx;
  float w = w1[c];
  float A = w * g1[c] * rsqrtf(w * w * vx + 1e-5f);
  aff[c] = A;
  aff[8 + c] = be1[c] - A * mx;
}

// ---------------------- dtransform conv2: 8->32, 5x5, s4, p2, t1 on the fly
__global__ void k_conv2(const float* __restrict__ dx, const float* __restrict__ aff,
                        const float* __restrict__ w2, const float* __restrict__ b2,
                        float* __restrict__ out) {
  long i = (long)blockIdx.x * blockDim.x + threadIdx.x;
  long n = (long)B_ * 32 * 96 * 96;
  if (i >= n) return;
  int ow = (int)(i % 96);
  int oh = (int)((i / 96) % 96);
  int co = (int)((i / (96 * 96)) % 32);
  int b  = (int)(i / ((long)96 * 96 * 32));
  float s = b2[co];
  const float* xb = dx + (long)b * 384 * 384;
  const float* wb = w2 + co * 8 * 25;
  for (int ci = 0; ci < 8; ci++) {
    float A = aff[ci], C = aff[8 + ci];
    for (int kh = 0; kh < 5; kh++) {
      int ih = 4 * oh + kh - 2;
      if (ih < 0 || ih >= 384) continue;
      for (int kw = 0; kw < 5; kw++) {
        int iw = 4 * ow + kw - 2;
        if (iw < 0 || iw >= 384) continue;
        float t = A * xb[ih * 384 + iw] + C;
        t = t > 0.f ? t : 0.f;
        s += wb[(ci * 5 + kh) * 5 + kw] * t;
      }
    }
  }
  out[i] = s;
}

// ---------------- generic 5x5 stride-2 pad-2 direct conv (compile-time shape)
template <int CIN, int IH, int COUT, int OH>
__global__ void k_conv5(const float* __restrict__ in, const float* __restrict__ w,
                        const float* __restrict__ bias, float* __restrict__ out) {
  long i = (long)blockIdx.x * blockDim.x + threadIdx.x;
  long n = (long)B_ * COUT * OH * OH;
  if (i >= n) return;
  int ow = (int)(i % OH);
  int oh = (int)((i / OH) % OH);
  int co = (int)((i / (OH * OH)) % COUT);
  int b  = (int)(i / ((long)OH * OH * COUT));
  float s = bias[co];
  for (int ci = 0; ci < CIN; ci++) {
    const float* xb = in + ((long)b * CIN + ci) * IH * IH;
    const float* wb = w + ((long)co * CIN + ci) * 25;
    for (int kh = 0; kh < 5; kh++) {
      int ih = 2 * oh + kh - 2;
      if (ih < 0 || ih >= IH) continue;
      for (int kw = 0; kw < 5; kw++) {
        int iw = 2 * ow + kw - 2;
        if (iw < 0 || iw >= IH) continue;
        s += wb[kh * 5 + kw] * xb[ih * IH + iw];
      }
    }
  }
  out[i] = s;
}

// ---------------------------------------------- BN batch stats (block/channel)
template <int C, int HW>
__global__ void k_bnstats(const float* __restrict__ x, float* __restrict__ mean,
                          float* __restrict__ var) {
  int c = blockIdx.x;
  __shared__ float ss[256], sq[256];
  float s = 0.f, q = 0.f;
  constexpr int N = B_ * HW;
  for (int j = threadIdx.x; j < N; j += 256) {
    int b = j / HW, r = j % HW;
    float v = x[((long)b * C + c) * HW + r];
    s += v; q += v * v;
  }
  ss[threadIdx.x] = s; sq[threadIdx.x] = q; __syncthreads();
  for (int o = 128; o > 0; o >>= 1) {
    if (threadIdx.x < o) { ss[threadIdx.x] += ss[threadIdx.x + o]; sq[threadIdx.x] += sq[threadIdx.x + o]; }
    __syncthreads();
  }
  if (threadIdx.x == 0) {
    float m = ss[0] / N;
    mean[c] = m;
    var[c] = sq[0] / N - m * m;
  }
}

template <int C, int HW>
__global__ void k_bnrelu(float* __restrict__ x, const float* __restrict__ mean,
                         const float* __restrict__ var, const float* __restrict__ g,
                         const float* __restrict__ be) {
  long i = (long)blockIdx.x * blockDim.x + threadIdx.x;
  long n = (long)B_ * C * HW;
  if (i >= n) return;
  int c = (int)((i / HW) % C);
  float v = (x[i] - mean[c]) * rsqrtf(var[c] + 1e-5f) * g[c] + be[c];
  x[i] = v > 0.f ? v : 0.f;
}

// BN+ReLU + convert to f16 NHWC (GEMM activation layout), HW = 576
template <int C>
__global__ void k_bnrelu16(const float* __restrict__ x, const float* __restrict__ mean,
                           const float* __restrict__ var, const float* __restrict__ g,
                           const float* __restrict__ be, _Float16* __restrict__ o) {
  long i = (long)blockIdx.x * blockDim.x + threadIdx.x;
  long n = (long)B_ * C * HW576;
  if (i >= n) return;
  int hw = (int)(i % HW576);
  int c  = (int)((i / HW576) % C);
  int b  = (int)(i / ((long)HW576 * C));
  float v = (x[i] - mean[c]) * rsqrtf(var[c] + 1e-5f) * g[c] + be[c];
  v = v > 0.f ? v : 0.f;
  o[((long)b * HW576 + hw) * C + c] = (_Float16)v;
}

// concat(d4, x_feat) -> f16 NHWC, C = 320
__global__ void k_pack_in16(const float* __restrict__ d4, const float* __restrict__ xf,
                            _Float16* __restrict__ o) {
  long i = (long)blockIdx.x * blockDim.x + threadIdx.x;
  long n = (long)B_ * HW576 * 320;
  if (i >= n) return;
  int c  = (int)(i % 320);
  int hw = (int)((i / 320) % HW576);
  int b  = (int)(i / ((long)320 * HW576));
  float v = (c < 64) ? d4[((long)b * 64 + c) * HW576 + hw]
                     : xf[((long)b * 256 + (c - 64)) * HW576 + hw];
  o[i] = (_Float16)v;
}

// pack OIHW f32 weights -> n-major [co][k] f16,  k = tap*Cin + ci, tap = kh*3+kw
__global__ void k_packW(const float* __restrict__ w, int Cout, int Cin, int taps,
                        _Float16* __restrict__ o) {
  long i = (long)blockIdx.x * blockDim.x + threadIdx.x;
  long n = (long)taps * Cin * Cout;
  if (i >= n) return;
  int co = (int)(i % Cout);
  long k = i / Cout;
  int ci = (int)(k % Cin);
  int tap = (int)(k / Cin);
  o[(long)co * (taps * Cin) + k] = (_Float16)w[((long)co * Cin + ci) * taps + tap];
}

// ---------------------------------------------------------------------------
// Implicit-GEMM conv via V_WMMA_F32_16X16X32_F16 (wave32).
//   M = B*24*24 = 18432 pixels, N = COUT, K = TAPS*CIN.
//   Block: 256 threads = 8 waves; tile 256(M) x 32(N).
//   Each wave owns 32x32 -> 4 accumulators, 2 A-frags x 2 B-frags per K-step
//   (4 WMMA per 8 fragment loads). A staged in LDS row-major so a fragment is
//   two ds_load_b128; weights are n-major in global (L2-resident, 1.5 MB) so
//   B fragments are two global_load_b128 per lane — no LDS for B at all.
// ---------------------------------------------------------------------------
template <int CIN, int TAPS, int COUT>
__global__ __launch_bounds__(256) void k_wmma_conv(
    const _Float16* __restrict__ inNHWC, const _Float16* __restrict__ Wp,
    const float* __restrict__ bias, float* __restrict__ out) {
  constexpr int KTOT = TAPS * CIN;
  constexpr int NK = KTOT / 32;
  __shared__ _Float16 As[256 * 32];          // 16 KB staging tile

  const int tid = threadIdx.x;
  const int wave = tid >> 5, lane = tid & 31;
  const int half = lane >> 4, r = lane & 15;
  const int m0 = blockIdx.x * 256;
  const int n0 = blockIdx.y * 32;

  // staging coords: this thread stages the full 32-wide K chunk of row m0+tid
  const int m = m0 + tid;
  const int b = m / HW576;
  const int hw = m % HW576;
  const int h = hw / 24, w = hw % 24;

  const int am0 = 32 * wave + r;             // LDS row of A-frag 0 (frag 1: +16)
  const _Float16* wrow0 = Wp + (long)(n0 + r) * KTOT;        // B col ni=0
  const _Float16* wrow1 = Wp + (long)(n0 + 16 + r) * KTOT;   // B col ni=1

  v8f acc00 = {}, acc01 = {}, acc10 = {}, acc11 = {};
  const v8h zero8 = {};

  for (int s = 0; s < NK; s++) {
    const int k0 = s * 32;
    const int tap = k0 / CIN;                // CIN % 32 == 0: constant in chunk
    const int cb = k0 - tap * CIN;
    int dh = 0, dw = 0;
    if (TAPS == 9) { dh = tap / 3 - 1; dw = tap % 3 - 1; }
    const int ih = h + dh, iw = w + dw;
    const bool inb = (ih >= 0) && (ih < 24) && (iw >= 0) && (iw < 24);
    const _Float16* src = inNHWC + (((long)b * 24 + ih) * 24 + iw) * CIN + cb;
#pragma unroll
    for (int q = 0; q < 4; q++) {            // 4x b128 global -> 4x b128 ds
      v8h v = inb ? *reinterpret_cast<const v8h*>(src + 8 * q) : zero8;
      *reinterpret_cast<v8h*>(&As[tid * 32 + 8 * q]) = v;
    }
    __syncthreads();

    // B fragments: K = k0 + {8h..8h+7} then {16+8h..23+8h} in n-major weights
    v8h bl0 = *reinterpret_cast<const v8h*>(wrow0 + k0 + 8 * half);
    v8h bh0 = *reinterpret_cast<const v8h*>(wrow0 + k0 + 16 + 8 * half);
    v8h bl1 = *reinterpret_cast<const v8h*>(wrow1 + k0 + 8 * half);
    v8h bh1 = *reinterpret_cast<const v8h*>(wrow1 + k0 + 16 + 8 * half);
    v16h bf0 = __builtin_shufflevector(bl0, bh0, 0, 1, 2, 3, 4, 5, 6, 7,
                                       8, 9, 10, 11, 12, 13, 14, 15);
    v16h bf1 = __builtin_shufflevector(bl1, bh1, 0, 1, 2, 3, 4, 5, 6, 7,
                                       8, 9, 10, 11, 12, 13, 14, 15);
    // A fragments: K = {8h..8h+7} then {16+8h..23+8h}, both b128-contiguous
    v8h al0 = *reinterpret_cast<const v8h*>(&As[am0 * 32 + 8 * half]);
    v8h ah0 = *reinterpret_cast<const v8h*>(&As[am0 * 32 + 16 + 8 * half]);
    v8h al1 = *reinterpret_cast<const v8h*>(&As[(am0 + 16) * 32 + 8 * half]);
    v8h ah1 = *reinterpret_cast<const v8h*>(&As[(am0 + 16) * 32 + 16 + 8 * half]);
    v16h af0 = __builtin_shufflevector(al0, ah0, 0, 1, 2, 3, 4, 5, 6, 7,
                                       8, 9, 10, 11, 12, 13, 14, 15);
    v16h af1 = __builtin_shufflevector(al1, ah1, 0, 1, 2, 3, 4, 5, 6, 7,
                                       8, 9, 10, 11, 12, 13, 14, 15);

    acc00 = __builtin_amdgcn_wmma_f32_16x16x32_f16(false, af0, false, bf0, (short)0, acc00, false, false);
    acc01 = __builtin_amdgcn_wmma_f32_16x16x32_f16(false, af0, false, bf1, (short)0, acc01, false, false);
    acc10 = __builtin_amdgcn_wmma_f32_16x16x32_f16(false, af1, false, bf0, (short)0, acc10, false, false);
    acc11 = __builtin_amdgcn_wmma_f32_16x16x32_f16(false, af1, false, bf1, (short)0, acc11, false, false);

    if (s + 1 < NK)                          // gfx1250 prefetch: next B chunk
      __builtin_prefetch(wrow0 + k0 + 32, 0, 1);
    __syncthreads();
  }

  // epilogue: D VGPR rr -> local M = rr + 8*half, N = lane%16; +bias, NCHW f32
  const float bv0 = bias[n0 + r];
  const float bv1 = bias[n0 + 16 + r];
#pragma unroll
  for (int rr = 0; rr < 8; rr++) {
    int mm0 = m0 + 32 * wave + rr + 8 * half;
    int mm1 = mm0 + 16;
    int bb0 = mm0 / HW576, hh0 = mm0 % HW576;
    int bb1 = mm1 / HW576, hh1 = mm1 % HW576;
    out[((long)bb0 * COUT + (n0 + r)) * HW576 + hh0]      = acc00[rr] + bv0;
    out[((long)bb0 * COUT + (n0 + 16 + r)) * HW576 + hh0] = acc01[rr] + bv1;
    out[((long)bb1 * COUT + (n0 + r)) * HW576 + hh1]      = acc10[rr] + bv0;
    out[((long)bb1 * COUT + (n0 + 16 + r)) * HW576 + hh1] = acc11[rr] + bv1;
  }
}

// ------------------------------------------ gaussian depth dist + softmax(D)
__global__ void k_gauss(const float* __restrict__ dx, float* __restrict__ dd) {
  int t = blockIdx.x * blockDim.x + threadIdx.x;
  if (t >= B_ * HW576) return;
  int b = t / HW576, hw = t % HW576, h = hw / 24, w = hw % 24;
  const float* xb = dx + (long)b * 384 * 384;
  float s = 0.f;
  for (int i = 0; i < 16; i++)
    for (int j = 0; j < 16; j++)
      s += xb[(h * 16 + i) * 384 + (w * 16 + j)];
  float avg = s * (1.0f / 256.0f);
  float m = floorf(avg * 0.001f * 118.0f);
  const float coef = 0.3989422804014327f * 0.5f;   // (1/sqrt(2pi))*sigma
  float sum = 0.f;
  for (int d = 0; d < D_; d++) {
    float dm = (float)d - m;
    float g = coef * expf(-dm * dm * 0.125f);      // exp(-(d-m)^2/2 * s^2)
    sum += expf(g);
  }
  float inv = 1.0f / sum;
  for (int d = 0; d < D_; d++) {
    float dm = (float)d - m;
    float g = coef * expf(-dm * dm * 0.125f);
    dd[((long)b * D_ + d) * HW576 + hw] = expf(g) * inv;
  }
}

// ----------------------------------- splat: outer product + scatter into BEV
__global__ __launch_bounds__(64) void k_splat(const int* __restrict__ vidx,
                                              const float* __restrict__ dd,
                                              const float* __restrict__ x,
                                              float* __restrict__ out) {
  long blk = blockIdx.x;
  int hw = (int)(blk % HW576);
  long t = blk / HW576;
  int d = (int)(t % D_);
  int b = (int)(t / D_);
  int vi = vidx[((long)b * D_ + d) * HW576 + hw];
  if (vi < 0) return;
  int vy = vi & 255;
  if (vy < 50) return;                  // bev[..., 50:] slice
  int vx = vi >> 8;
  float dv = dd[((long)b * D_ + d) * HW576 + hw];
  int c = threadIdx.x;                  // 64 channels
  float val = dv * x[((long)b * 64 + c) * HW576 + hw];
  atomicAdd(&out[(((long)b * 64 + c) * NX_ + vx) * NYOUT + (vy - 50)], val);
}

// ===========================================================================
extern "C" void kernel_launch(void* const* d_in, const int* in_sizes, int n_in,
                              void* d_out, int out_size, void* d_ws, size_t ws_size,
                              hipStream_t stream) {
  const float* x_feat  = (const float*)d_in[0];
  const float* intrins = (const float*)d_in[1];
  const float* depth_x = (const float*)d_in[2];
  const float* dt_w1 = (const float*)d_in[3];
  const float* dt_g1 = (const float*)d_in[5];
  const float* dt_be1 = (const float*)d_in[6];
  const float* dt_w2 = (const float*)d_in[7];
  const float* dt_b2 = (const float*)d_in[8];
  const float* dt_g2 = (const float*)d_in[9];
  const float* dt_be2 = (const float*)d_in[10];
  const float* dt_w3 = (const float*)d_in[11];
  const float* dt_b3 = (const float*)d_in[12];
  const float* dt_g3 = (const float*)d_in[13];
  const float* dt_be3 = (const float*)d_in[14];
  const float* dt_w4 = (const float*)d_in[15];
  const float* dt_b4 = (const float*)d_in[16];
  const float* dt_g4 = (const float*)d_in[17];
  const float* dt_be4 = (const float*)d_in[18];
  const float* dn_w1 = (const float*)d_in[19];
  const float* dn_b1 = (const float*)d_in[20];
  const float* dn_g1 = (const float*)d_in[21];
  const float* dn_be1 = (const float*)d_in[22];
  const float* dn_w2 = (const float*)d_in[23];
  const float* dn_b2 = (const float*)d_in[24];
  const float* dn_g2 = (const float*)d_in[25];
  const float* dn_be2 = (const float*)d_in[26];
  const float* dn_w3 = (const float*)d_in[27];
  const float* dn_b3 = (const float*)d_in[28];

  float* out = (float*)d_out;

  // workspace carve-up (256B aligned regions; ~89 MB total with reuse)
  char* ws = (char*)d_ws;
  size_t off = 0;
  auto alloc = [&](size_t bytes) -> char* {
    char* p = ws + off;
    off = (off + bytes + 255) & ~(size_t)255;
    return p;
  };
  float*    comb  = (float*)alloc(B_ * 9 * 4);
  float*    stat0 = (float*)alloc(2 * 4);
  float*    aff1  = (float*)alloc(16 * 4);
  float*    bnm   = (float*)alloc(256 * 4);          // reused per stage
  float*    bnv   = (float*)alloc(256 * 4);
  float*    t2reg = (float*)alloc((size_t)B_ * 32 * 96 * 96 * 4);  // t2 / dn1raw / dn2raw
  float*    t3    = (float*)alloc((size_t)B_ * 32 * 48 * 48 * 4);
  float*    d4    = (float*)alloc((size_t)B_ * 64 * HW576 * 4);
  _Float16* act16 = (_Float16*)alloc((size_t)B_ * HW576 * 320 * 2); // in16a/b/c
  float*    xout  = (float*)alloc((size_t)B_ * 64 * HW576 * 4);
  float*    dd    = (float*)alloc((size_t)B_ * D_ * HW576 * 4);
  int*      vidx  = (int*)alloc((size_t)B_ * D_ * HW576 * 4);
  _Float16* wp1   = (_Float16*)alloc((size_t)2880 * 256 * 2);
  _Float16* wp2   = (_Float16*)alloc((size_t)2304 * 256 * 2);
  _Float16* wp3   = (_Float16*)alloc((size_t)256 * 64 * 2);
  (void)ws_size; (void)in_sizes; (void)n_in;

  // 0. zero output BEV + stat accumulators
  k_zero_f32<<<8192, 256, 0, stream>>>(out, (long)out_size);
  k_zero_f32<<<1, 256, 0, stream>>>(stat0, 2);

  // 1. geometry -> voxel indices
  k_combine<<<1, 32, 0, stream>>>(intrins, comb);
  {
    long n = (long)B_ * D_ * HW576;
    k_voxel<<<(int)((n + 255) / 256), 256, 0, stream>>>(comb, vidx);
  }

  // 2. dtransform: folded conv1+BN, then direct convs + BN
  k_stat0<<<512, 256, 0, stream>>>(depth_x, (long)B_ * 384 * 384, stat0);
  k_aff1<<<1, 8, 0, stream>>>(stat0, dt_w1, dt_g1, dt_be1, aff1);
  {
    long n = (long)B_ * 32 * 96 * 96;
    k_conv2<<<(int)((n + 255) / 256), 256, 0, stream>>>(depth_x, aff1, dt_w2, dt_b2, t2reg);
    k_bnstats<32, 96 * 96><<<32, 256, 0, stream>>>(t2reg, bnm, bnv);
    k_bnrelu<32, 96 * 96><<<(int)((n + 255) / 256), 256, 0, stream>>>(t2reg, bnm, bnv, dt_g2, dt_be2);
  }
  {
    long n = (long)B_ * 32 * 48 * 48;
    k_conv5<32, 96, 32, 48><<<(int)((n + 255) / 256), 256, 0, stream>>>(t2reg, dt_w3, dt_b3, t3);
    k_bnstats<32, 48 * 48><<<32, 256, 0, stream>>>(t3, bnm, bnv);
    k_bnrelu<32, 48 * 48><<<(int)((n + 255) / 256), 256, 0, stream>>>(t3, bnm, bnv, dt_g3, dt_be3);
  }
  {
    long n = (long)B_ * 64 * HW576;
    k_conv5<32, 48, 64, 24><<<(int)((n + 255) / 256), 256, 0, stream>>>(t3, dt_w4, dt_b4, d4);
    k_bnstats<64, HW576><<<64, 256, 0, stream>>>(d4, bnm, bnv);
    k_bnrelu<64, HW576><<<(int)((n + 255) / 256), 256, 0, stream>>>(d4, bnm, bnv, dt_g4, dt_be4);
  }

  // 3. pack weights + activations for WMMA implicit GEMM
  {
    long n = (long)2880 * 256;
    k_packW<<<(int)((n + 255) / 256), 256, 0, stream>>>(dn_w1, 256, 320, 9, wp1);
    n = (long)2304 * 256;
    k_packW<<<(int)((n + 255) / 256), 256, 0, stream>>>(dn_w2, 256, 256, 9, wp2);
    n = (long)256 * 64;
    k_packW<<<(int)((n + 255) / 256), 256, 0, stream>>>(dn_w3, 64, 256, 1, wp3);
    n = (long)B_ * HW576 * 320;
    k_pack_in16<<<(int)((n + 255) / 256), 256, 0, stream>>>(d4, x_feat, act16);
  }

  // 4. depthnet: WMMA GEMM convs + BN  (M tile 256 -> grid.x = 18432/256 = 72)
  k_wmma_conv<320, 9, 256><<<dim3(72, 8), 256, 0, stream>>>(act16, wp1, dn_b1, t2reg);
  {
    long n = (long)B_ * 256 * HW576;
    k_bnstats<256, HW576><<<256, 256, 0, stream>>>(t2reg, bnm, bnv);
    k_bnrelu16<256><<<(int)((n + 255) / 256), 256, 0, stream>>>(t2reg, bnm, bnv, dn_g1, dn_be1, act16);
  }
  k_wmma_conv<256, 9, 256><<<dim3(72, 8), 256, 0, stream>>>(act16, wp2, dn_b2, t2reg);
  {
    long n = (long)B_ * 256 * HW576;
    k_bnstats<256, HW576><<<256, 256, 0, stream>>>(t2reg, bnm, bnv);
    k_bnrelu16<256><<<(int)((n + 255) / 256), 256, 0, stream>>>(t2reg, bnm, bnv, dn_g2, dn_be2, act16);
  }
  k_wmma_conv<256, 1, 64><<<dim3(72, 2), 256, 0, stream>>>(act16, wp3, dn_b3, xout);

  // 5. gaussian depth distribution
  k_gauss<<<(B_ * HW576 + 255) / 256, 256, 0, stream>>>(depth_x, dd);

  // 6. outer-product splat + scatter-add into sliced BEV
  {
    long nblk = (long)B_ * D_ * HW576;
    k_splat<<<(int)nblk, 64, 0, stream>>>(vidx, dd, xout, out);
  }
}
